// GatedMultiHeadAttention_31576599561042
// MI455X (gfx1250) — matrix operand order repaired and checked by reading, not compiled, not run on hardware
//
#include <hip/hip_runtime.h>
#include <math.h>

// ---------------- problem constants ----------------
#define BB 4
#define SS 2048
#define DD 1024
#define HH 16
#define DH 64
#define MM (BB*SS)   // 8192 rows

typedef __attribute__((ext_vector_type(16))) _Float16 v16h;
typedef __attribute__((ext_vector_type(8)))  _Float16 v8h;
typedef __attribute__((ext_vector_type(4)))  _Float16 v4h;
typedef __attribute__((ext_vector_type(8)))  float    v8f;
typedef __attribute__((ext_vector_type(4)))  float    fl4;

__device__ __forceinline__ v8f wmma_f16(v16h a, v16h b, v8f c) {
  return __builtin_amdgcn_wmma_f32_16x16x32_f16(false, a, false, b, (short)0, c, false, false);
}

__device__ __forceinline__ float gelu_exact(float x) {
  return 0.5f * x * (1.0f + erff(x * 0.70710678118654752f));
}

__device__ __forceinline__ float wave_red_sum(float v) {
  #pragma unroll
  for (int o = 16; o > 0; o >>= 1) v += __shfl_xor(v, o, 32);
  return v;
}

// =====================================================================
// Kernel 1: projection GEMM  out[m,n] = X[m,:] . W[n,:] + bias[n]
// MODE 0: store f16 into head layout [B,H,S,Dh]   (q,k)
// MODE 1: exact GELU, store f32 row-major [B,S,D] (v,g)
// grid (64, 8), block 256 (8 waves). Block tile 128x128, K-step 32.
// Software-pipelined: next K-tile global loads overlap with WMMA.
// =====================================================================
template <int MODE>
__global__ __launch_bounds__(256)
void proj_gemm(const float* __restrict__ X, const float* __restrict__ W,
               const float* __restrict__ bias,
               _Float16* __restrict__ outH, float* __restrict__ outF) {
  __shared__ _Float16 tileA[128][40];   // [m][k]
  __shared__ _Float16 tileB[128][40];   // [n][k] = W[n][k]

  const int t   = threadIdx.x;
  const int w   = t >> 5;
  const int L   = t & 31;
  const int ln  = L & 15;
  const int hi  = L >> 4;
  const int mBlk = blockIdx.x * 128;
  const int nBlk = blockIdx.y * 128;
  const int m0w  = (w >> 1) * 32;      // wave: 32 rows x 64 cols
  const int n0w  = (w & 1) * 64;
  const int r0 = t >> 3, cc = (t & 7) * 4;

  v8f c[2][4];
  #pragma unroll
  for (int i = 0; i < 2; ++i)
    #pragma unroll
    for (int j = 0; j < 4; ++j) c[i][j] = {};

  // pipeline registers: raw f32 tile data for current store stage
  fl4 ra[4], rb[4];
  #pragma unroll
  for (int u = 0; u < 4; ++u) {
    const int rr = r0 + 32 * u;
    ra[u] = *reinterpret_cast<const fl4*>(X + (size_t)(mBlk + rr) * DD + cc);
    rb[u] = *reinterpret_cast<const fl4*>(W + (size_t)(nBlk + rr) * DD + cc);
  }

  for (int kb = 0; kb < DD; kb += 32) {
    // store stage (waits on this tile's loads)
    #pragma unroll
    for (int u = 0; u < 4; ++u) {
      const int rr = r0 + 32 * u;
      v4h ha = { (_Float16)ra[u].x, (_Float16)ra[u].y, (_Float16)ra[u].z, (_Float16)ra[u].w };
      *reinterpret_cast<v4h*>(&tileA[rr][cc]) = ha;
      v4h hb = { (_Float16)rb[u].x, (_Float16)rb[u].y, (_Float16)rb[u].z, (_Float16)rb[u].w };
      *reinterpret_cast<v4h*>(&tileB[rr][cc]) = hb;
    }
    __syncthreads();

    // issue next tile's global loads (overlap with WMMA below)
    if (kb + 32 < DD) {
      #pragma unroll
      for (int u = 0; u < 4; ++u) {
        const int rr = r0 + 32 * u;
        ra[u] = *reinterpret_cast<const fl4*>(X + (size_t)(mBlk + rr) * DD + kb + 32 + cc);
        rb[u] = *reinterpret_cast<const fl4*>(W + (size_t)(nBlk + rr) * DD + kb + 32 + cc);
      }
      if (kb + 64 < DD) {   // pull tile-after-next toward GL2 (global_prefetch_b8)
        __builtin_prefetch(X + (size_t)(mBlk + r0) * DD + kb + 64 + cc, 0, 0);
        __builtin_prefetch(W + (size_t)(nBlk + r0) * DD + kb + 64 + cc, 0, 0);
      }
    }

    // A fragments (2): lane row m, K = {e+8hi, 16+e+8hi}
    v16h af[2];
    #pragma unroll
    for (int i = 0; i < 2; ++i) {
      const int m = m0w + 16 * i + ln;
      reinterpret_cast<v8h*>(&af[i])[0] = *reinterpret_cast<const v8h*>(&tileA[m][8 * hi]);
      reinterpret_cast<v8h*>(&af[i])[1] = *reinterpret_cast<const v8h*>(&tileA[m][16 + 8 * hi]);
    }
    // B fragments (4): lane col n, K = {e+8hi, 16+e+8hi}
    v16h bf[4];
    #pragma unroll
    for (int j = 0; j < 4; ++j) {
      const int n = n0w + 16 * j + ln;
      reinterpret_cast<v8h*>(&bf[j])[0] = *reinterpret_cast<const v8h*>(&tileB[n][8 * hi]);
      reinterpret_cast<v8h*>(&bf[j])[1] = *reinterpret_cast<const v8h*>(&tileB[n][16 + 8 * hi]);
    }
    #pragma unroll
    for (int i = 0; i < 2; ++i)
      #pragma unroll
      for (int j = 0; j < 4; ++j)
        c[i][j] = wmma_f16(af[i], bf[j], c[i][j]);

    __syncthreads();   // LDS reads done before next store stage
  }

  // epilogue: C layout  (m = r + 8*hi, n = lane%16)
  #pragma unroll
  for (int i = 0; i < 2; ++i) {
    #pragma unroll
    for (int j = 0; j < 4; ++j) {
      const int n = nBlk + n0w + 16 * j + ln;
      const float bv = bias[n];
      #pragma unroll
      for (int r = 0; r < 8; ++r) {
        const int m = mBlk + m0w + 16 * i + r + 8 * hi;
        float val = c[i][j][r] + bv;
        if (MODE == 0) {
          const int bidx = m >> 11, s = m & (SS - 1);
          const int h = n >> 6, dh = n & 63;
          outH[(((size_t)(bidx * HH + h) * SS) + s) * DH + dh] = (_Float16)val;
        } else {
          outF[(size_t)m * DD + n] = gelu_exact(val);
        }
      }
    }
  }
}

// =====================================================================
// Kernel 2: LayerNorm (unbiased var, eps=1e-8) on vp rows, write V
// transposed per head: vhT[b,h,dh,s] (f16).  grid 8192, block 256.
// =====================================================================
__global__ __launch_bounds__(256)
void ln_kernel(const float* __restrict__ vp, const float* __restrict__ ln_a,
               const float* __restrict__ ln_b, _Float16* __restrict__ vhT) {
  __shared__ float warr[8], warr2[8], mv[2];
  const int row = blockIdx.x;
  const int b = row >> 11, s = row & (SS - 1);
  const int t = threadIdx.x, w = t >> 5, L = t & 31;

  float x[4], sum = 0.f, sq = 0.f;
  #pragma unroll
  for (int j = 0; j < 4; ++j) {
    x[j] = vp[(size_t)row * DD + t + 256 * j];
    sum += x[j]; sq += x[j] * x[j];
  }
  sum = wave_red_sum(sum); sq = wave_red_sum(sq);
  if (L == 0) { warr[w] = sum; warr2[w] = sq; }
  __syncthreads();
  if (t == 0) {
    float S1 = 0.f, S2 = 0.f;
    #pragma unroll
    for (int i = 0; i < 8; ++i) { S1 += warr[i]; S2 += warr2[i]; }
    const float mu = S1 / (float)DD;
    const float var = (S2 - (float)DD * mu * mu) / (float)(DD - 1);
    mv[0] = mu; mv[1] = rsqrtf(var + 1e-8f);
  }
  __syncthreads();
  const float mu = mv[0], rstd = mv[1];
  #pragma unroll
  for (int j = 0; j < 4; ++j) {
    const int d = t + 256 * j;
    const float y = (x[j] - mu) * rstd * ln_a[d] + ln_b[d];
    const int h = d >> 6, dh = d & 63;
    vhT[(((size_t)(b * HH + h) * DH) + dh) * SS + s] = (_Float16)y;
  }
}

// =====================================================================
// Kernel 3: attention for one (b, h, 16-query block).
// grid (128, 16, 4), block 128 (4 waves). Raw scores -> a (L2-resident),
// softmax passes re-read from L2, A*V reads probs from L2.
// =====================================================================
__global__ __launch_bounds__(128)
void attn_kernel(const _Float16* __restrict__ qh, const _Float16* __restrict__ kh,
                 const _Float16* __restrict__ vhT, const float* __restrict__ gp,
                 _Float16* __restrict__ og, float* __restrict__ a_out) {
  __shared__ float redbuf[128][8];
  __shared__ float psum[4][16];
  __shared__ float rowmax[16], rowinv[16];
  __shared__ float opart[4][16][64];

  const int t = threadIdx.x, w = t >> 5, L = t & 31;
  const int ln = L & 15, hi = L >> 4;
  const int q0 = blockIdx.x * 16;
  const int h = blockIdx.y, b = blockIdx.z;
  const size_t headOff = (size_t)(b * HH + h);
  const _Float16* qB = qh + headOff * SS * DH;
  const _Float16* kB = kh + headOff * SS * DH;
  const _Float16* vT = vhT + headOff * DH * SS;
  float* aB = a_out + headOff * (size_t)SS * SS + (size_t)q0 * SS;

  // Q fragments (constant across key tiles): row = q0+ln
  v16h qa0, qa1;
  {
    const _Float16* qr = qB + (size_t)(q0 + ln) * DH;
    reinterpret_cast<v8h*>(&qa0)[0] = *reinterpret_cast<const v8h*>(qr + 8 * hi);
    reinterpret_cast<v8h*>(&qa0)[1] = *reinterpret_cast<const v8h*>(qr + 16 + 8 * hi);
    reinterpret_cast<v8h*>(&qa1)[0] = *reinterpret_cast<const v8h*>(qr + 32 + 8 * hi);
    reinterpret_cast<v8h*>(&qa1)[1] = *reinterpret_cast<const v8h*>(qr + 48 + 8 * hi);
  }

  // ---- phase 1: raw scores + running max ----
  const int ntiles = q0 / 16 + 1;           // key tiles intersecting causal region
  float mx[8];
  #pragma unroll
  for (int r = 0; r < 8; ++r) mx[r] = -INFINITY;

  for (int kt = w; kt < ntiles; kt += 4) {
    const int key0 = kt * 16;
    const _Float16* kr = kB + (size_t)(key0 + ln) * DH;
    if (kt + 4 < ntiles)                    // prefetch this wave's next key tile
      __builtin_prefetch(kB + (size_t)(key0 + 64 + ln) * DH, 0, 0);
    v16h bf0, bf1;
    reinterpret_cast<v8h*>(&bf0)[0] = *reinterpret_cast<const v8h*>(kr + 8 * hi);
    reinterpret_cast<v8h*>(&bf0)[1] = *reinterpret_cast<const v8h*>(kr + 16 + 8 * hi);
    reinterpret_cast<v8h*>(&bf1)[0] = *reinterpret_cast<const v8h*>(kr + 32 + 8 * hi);
    reinterpret_cast<v8h*>(&bf1)[1] = *reinterpret_cast<const v8h*>(kr + 48 + 8 * hi);
    v8f c = {};
    c = wmma_f16(qa0, bf0, c);
    c = wmma_f16(qa1, bf1, c);
    #pragma unroll
    for (int r = 0; r < 8; ++r) {
      const int m = r + 8 * hi;
      const int key = key0 + ln;
      float val = c[r] * 0.125f;
      if (key > q0 + m) val = -INFINITY;    // causal mask
      aB[(size_t)m * SS + key] = val;
      mx[r] = fmaxf(mx[r], val);
    }
  }
  #pragma unroll
  for (int r = 0; r < 8; ++r) redbuf[t][r] = mx[r];
  __threadfence_block();
  __syncthreads();
  if (t < 16) {
    const int m = t, him = m >> 3, r = m & 7;
    float mm = -INFINITY;
    for (int w2 = 0; w2 < 4; ++w2)
      for (int l2 = 0; l2 < 16; ++l2)
        mm = fmaxf(mm, redbuf[w2 * 32 + him * 16 + l2][r]);
    rowmax[m] = mm;
  }
  __syncthreads();

  // ---- phase 2a: exp-sums (read raw scores from L2) ----
  const int maxcol = q0 + 15;
  for (int m = 0; m < 16; ++m) {
    const float rmx = rowmax[m];
    float ts = 0.f;
    for (int cc = t; cc <= maxcol; cc += 128)
      ts += expf(aB[(size_t)m * SS + cc] - rmx);
    ts = wave_red_sum(ts);
    if (L == 0) psum[w][m] = ts;
  }
  __syncthreads();
  if (t < 16) rowinv[t] = 1.0f / (psum[0][t] + psum[1][t] + psum[2][t] + psum[3][t]);
  __syncthreads();

  // ---- phase 2b: write normalized probabilities (the `a` output) ----
  for (int m = 0; m < 16; ++m) {
    const float rmx = rowmax[m], rin = rowinv[m];
    for (int cc = t; cc < SS; cc += 128) {
      float p = 0.f;
      if (cc <= maxcol) p = expf(aB[(size_t)m * SS + cc] - rmx) * rin;
      aB[(size_t)m * SS + cc] = p;
    }
  }
  __threadfence_block();
  __syncthreads();

  // ---- phase 3: O = P * V  (A frags from a in L2, B frags from vT) ----
  v8f oc[4];
  #pragma unroll
  for (int j = 0; j < 4; ++j) oc[j] = {};
  const int nsteps = (q0 + 16 + 31) / 32;
  for (int st = w; st < nsteps; st += 4) {
    const int kb = st * 32;
    if (st + 4 < nsteps) {                  // prefetch next V strip
      __builtin_prefetch(vT + (size_t)ln * SS + kb + 128, 0, 0);
      __builtin_prefetch(aB + (size_t)ln * SS + kb + 128, 0, 0);
    }
    const float* pr1 = aB + (size_t)ln * SS + kb + 8 * hi;
    const float* pr2 = pr1 + 16;
    v16h pa;
    #pragma unroll
    for (int e = 0; e < 8; ++e) {
      pa[e]     = (_Float16)pr1[e];
      pa[8 + e] = (_Float16)pr2[e];
    }
    #pragma unroll
    for (int j = 0; j < 4; ++j) {
      const _Float16* vr = vT + (size_t)(j * 16 + ln) * SS + kb + 8 * hi;
      v16h vb;
      reinterpret_cast<v8h*>(&vb)[0] = *reinterpret_cast<const v8h*>(vr);
      reinterpret_cast<v8h*>(&vb)[1] = *reinterpret_cast<const v8h*>(vr + 16);
      oc[j] = wmma_f16(pa, vb, oc[j]);
    }
  }
  #pragma unroll
  for (int j = 0; j < 4; ++j)
    #pragma unroll
    for (int r = 0; r < 8; ++r)
      opart[w][r + 8 * hi][j * 16 + ln] = oc[j][r];
  __syncthreads();

  // reduce partials, apply gate, store og (f16, [B,S,D])
  for (int i = t; i < 16 * 64; i += 128) {
    const int m = i >> 6, n = i & 63;
    const float o = opart[0][m][n] + opart[1][m][n] + opart[2][m][n] + opart[3][m][n];
    const int srow = q0 + m;
    const size_t idx = ((size_t)b * SS + srow) * DD + h * DH + n;
    og[idx] = (_Float16)(o * gp[idx]);
  }
}

// =====================================================================
// Kernel 4: final GEMM  out[m,n] = OG[m,:] . Wo[n,:] + bo[n]  (f32 out)
// grid (64, 8), block 256.  Software-pipelined like proj_gemm.
// =====================================================================
__global__ __launch_bounds__(256)
void final_gemm(const _Float16* __restrict__ OG, const float* __restrict__ W,
                const float* __restrict__ bias, float* __restrict__ out) {
  __shared__ _Float16 tileA[128][40];
  __shared__ _Float16 tileB[128][40];

  const int t = threadIdx.x, w = t >> 5, L = t & 31;
  const int ln = L & 15, hi = L >> 4;
  const int mBlk = blockIdx.x * 128, nBlk = blockIdx.y * 128;
  const int m0w = (w >> 1) * 32, n0w = (w & 1) * 64;
  const int rA = t >> 1, cA = (t & 1) * 16;      // A: 16 halves/thread
  const int rB = t >> 3, cB = (t & 7) * 4;       // B: 16 f32/thread

  v8f c[2][4];
  #pragma unroll
  for (int i = 0; i < 2; ++i)
    #pragma unroll
    for (int j = 0; j < 4; ++j) c[i][j] = {};

  v16h haA;
  fl4 rb[4];
  haA = *reinterpret_cast<const v16h*>(OG + (size_t)(mBlk + rA) * DD + cA);
  #pragma unroll
  for (int u = 0; u < 4; ++u)
    rb[u] = *reinterpret_cast<const fl4*>(W + (size_t)(nBlk + rB + 32 * u) * DD + cB);

  for (int kb = 0; kb < DD; kb += 32) {
    *reinterpret_cast<v8h*>(&tileA[rA][cA])     = reinterpret_cast<v8h*>(&haA)[0];
    *reinterpret_cast<v8h*>(&tileA[rA][cA + 8]) = reinterpret_cast<v8h*>(&haA)[1];
    #pragma unroll
    for (int u = 0; u < 4; ++u) {
      const int rr = rB + 32 * u;
      v4h hb = { (_Float16)rb[u].x, (_Float16)rb[u].y, (_Float16)rb[u].z, (_Float16)rb[u].w };
      *reinterpret_cast<v4h*>(&tileB[rr][cB]) = hb;
    }
    __syncthreads();

    if (kb + 32 < DD) {
      haA = *reinterpret_cast<const v16h*>(OG + (size_t)(mBlk + rA) * DD + kb + 32 + cA);
      #pragma unroll
      for (int u = 0; u < 4; ++u)
        rb[u] = *reinterpret_cast<const fl4*>(W + (size_t)(nBlk + rB + 32 * u) * DD + kb + 32 + cB);
      if (kb + 64 < DD) {
        __builtin_prefetch(OG + (size_t)(mBlk + rA) * DD + kb + 64 + cA, 0, 0);
        __builtin_prefetch(W + (size_t)(nBlk + rB) * DD + kb + 64 + cB, 0, 0);
      }
    }

    v16h af[2];
    #pragma unroll
    for (int i = 0; i < 2; ++i) {
      const int m = m0w + 16 * i + ln;
      reinterpret_cast<v8h*>(&af[i])[0] = *reinterpret_cast<const v8h*>(&tileA[m][8 * hi]);
      reinterpret_cast<v8h*>(&af[i])[1] = *reinterpret_cast<const v8h*>(&tileA[m][16 + 8 * hi]);
    }
    v16h bf[4];
    #pragma unroll
    for (int j = 0; j < 4; ++j) {
      const int n = n0w + 16 * j + ln;
      reinterpret_cast<v8h*>(&bf[j])[0] = *reinterpret_cast<const v8h*>(&tileB[n][8 * hi]);
      reinterpret_cast<v8h*>(&bf[j])[1] = *reinterpret_cast<const v8h*>(&tileB[n][16 + 8 * hi]);
    }
    #pragma unroll
    for (int i = 0; i < 2; ++i)
      #pragma unroll
      for (int j = 0; j < 4; ++j)
        c[i][j] = wmma_f16(af[i], bf[j], c[i][j]);

    __syncthreads();
  }

  #pragma unroll
  for (int i = 0; i < 2; ++i)
    #pragma unroll
    for (int j = 0; j < 4; ++j) {
      const int n = nBlk + n0w + 16 * j + ln;
      const float bv = bias[n];
      #pragma unroll
      for (int r = 0; r < 8; ++r) {
        const int m = mBlk + m0w + 16 * i + r + 8 * hi;
        out[(size_t)m * DD + n] = c[i][j][r] + bv;
      }
    }
}

// =====================================================================
extern "C" void kernel_launch(void* const* d_in, const int* in_sizes, int n_in,
                              void* d_out, int out_size, void* d_ws, size_t ws_size,
                              hipStream_t stream) {
  (void)in_sizes; (void)n_in; (void)out_size; (void)ws_size;
  const float* q    = (const float*)d_in[0];
  const float* k    = (const float*)d_in[1];
  const float* v    = (const float*)d_in[2];
  const float* g    = (const float*)d_in[3];
  // d_in[4] = mask (causal, implemented analytically)
  const float* Wq   = (const float*)d_in[5];
  const float* bq   = (const float*)d_in[6];
  const float* Wk   = (const float*)d_in[7];
  const float* bk   = (const float*)d_in[8];
  const float* Wv   = (const float*)d_in[9];
  const float* bv   = (const float*)d_in[10];
  const float* Wg   = (const float*)d_in[11];
  const float* bg   = (const float*)d_in[12];
  const float* ln_a = (const float*)d_in[13];
  const float* ln_b = (const float*)d_in[14];
  const float* Wo   = (const float*)d_in[15];
  const float* bo   = (const float*)d_in[16];

  float* out = (float*)d_out;                       // o: [B,S,D] f32
  float* a_out = out + (size_t)BB * SS * DD;        // a: [B,H,S,S] f32

  // workspace carve-up
  char* ws = (char*)d_ws;
  const size_t NEL = (size_t)MM * DD;               // 8,388,608
  _Float16* qh  = (_Float16*)(ws);                  // f16 [B,H,S,Dh]
  _Float16* kh  = (_Float16*)(ws + NEL * 2);
  float*    vp  = (float*)   (ws + NEL * 4);        // f32 [B,S,D]
  float*    gp  = (float*)   (ws + NEL * 8);
  _Float16* vhT = (_Float16*)(ws + NEL * 12);       // f16 [B,H,Dh,S]
  _Float16* og  = (_Float16*)(ws + NEL * 14);       // f16 [B,S,D]

  dim3 gGemm(MM / 128, DD / 128);
  proj_gemm<0><<<gGemm, 256, 0, stream>>>(q, Wq, bq, qh, nullptr);
  proj_gemm<0><<<gGemm, 256, 0, stream>>>(k, Wk, bk, kh, nullptr);
  proj_gemm<1><<<gGemm, 256, 0, stream>>>(v, Wv, bv, nullptr, vp);
  proj_gemm<1><<<gGemm, 256, 0, stream>>>(g, Wg, bg, nullptr, gp);

  ln_kernel<<<MM, 256, 0, stream>>>(vp, ln_a, ln_b, vhT);

  attn_kernel<<<dim3(SS / 16, HH, BB), 128, 0, stream>>>(qh, kh, vhT, gp, og, a_out);

  final_gemm<<<gGemm, 256, 0, stream>>>(og, Wo, bo, out);
}
